// OutfitAwareFineGrainedCompatibility_61186104099402
// MI455X (gfx1250) — matrix operand order repaired and checked by reading, not compiled, not run on hardware
//
#include <hip/hip_runtime.h>
#include <math.h>

typedef __attribute__((ext_vector_type(16))) __bf16   v16bf;
typedef __attribute__((ext_vector_type(8)))  float    v8f;
typedef __attribute__((ext_vector_type(8)))  unsigned v8u;

#define Bm   512
#define Mm   16
#define Dm   512
#define Kk   8
#define Om   64
#define Hh   128
#define PPK  (Dm*Om + Om)     // 32832
#define NROW ((size_t)Kk*PPK) // 262656, fcp_W row stride
#define DC   16               // D-chunk size

// float -> bf16 bit helpers. pk_bf16: add rounding bias then one v_perm_b32
// to splice the two high halves (3 VALU per pair; RNE except exact ties).
__device__ __forceinline__ unsigned short bf16b(float x) {
  unsigned a = __float_as_uint(x);
  return (unsigned short)((a + 0x8000u) >> 16);
}
__device__ __forceinline__ unsigned pk_bf16(float lo, float hi) {
  unsigned a = __float_as_uint(lo) + 0x8000u;
  unsigned b = __float_as_uint(hi) + 0x8000u;
  return __builtin_amdgcn_perm(b, a, 0x07060302u);  // {hi_bf16, lo_bf16}
}
__device__ __forceinline__ float2 ld2(const float* p) { return *(const float2*)p; }

// ---------------- ctx = masked mean over M ----------------
__global__ void ctx_kernel(const float* __restrict__ item,
                           const int* __restrict__ mask,
                           float* __restrict__ ctx) {
  int b = blockIdx.x;
  float cnt = 0.f;
#pragma unroll
  for (int m = 0; m < Mm; ++m) cnt += (float)mask[b*Mm + m];
  cnt = fmaxf(cnt, 1e-6f);
  for (int d = threadIdx.x; d < Dm; d += blockDim.x) {
    float s = 0.f;
#pragma unroll
    for (int m = 0; m < Mm; ++m)
      s += (float)mask[b*Mm + m] * item[((size_t)b*Mm + m)*Dm + d];
    ctx[(size_t)b*Dm + d] = s / cnt;
  }
}

// ---------------- generic bf16-WMMA GEMM: C = act(A@B + bias) ----------------
template <int RELU>
__global__ void gemm_wmma(const float* __restrict__ A, const float* __restrict__ Bw,
                          const float* __restrict__ bias, float* __restrict__ C,
                          int Mdim, int Ndim, int Kdim) {
  int tid = threadIdx.x, wave = tid >> 5, lane = tid & 31;
  int half = lane >> 4, idx = lane & 15;
  int ntiles = Ndim >> 4;
  int tile = blockIdx.x * 8 + wave;
  if (tile >= (Mdim >> 4) * ntiles) return;   // wave-uniform
  int mt = tile / ntiles, nt = tile % ntiles;

  const float* arow = A + (size_t)(mt*16 + idx)*Kdim;
  const float* bcol = Bw + nt*16 + idx;

  v8f acc;
#pragma unroll
  for (int r = 0; r < 8; ++r) acc[r] = 0.f;

  for (int k0 = 0; k0 < Kdim; k0 += 32) {
    v8u ap, bp;
#pragma unroll
    for (int j = 0; j < 8; ++j) {
      int e = 2*j;
      int Ka = k0 + e + 8*half + ((e >= 8) ? 8 : 0);   // A 16x32 layout
      float2 av = ld2(&arow[Ka]);
      ap[j] = pk_bf16(av.x, av.y);
      int Kb = k0 + e + 8*half + ((e >= 8) ? 8 : 0);   // B 32x16 layout
      bp[j] = pk_bf16(bcol[(size_t)Kb*Ndim], bcol[(size_t)(Kb + 1)*Ndim]);
    }
    acc = __builtin_amdgcn_wmma_f32_16x16x32_bf16(false, __builtin_bit_cast(v16bf, ap),
                                                  false, __builtin_bit_cast(v16bf, bp),
                                                  (short)0, acc, false, false);
  }
#pragma unroll
  for (int r = 0; r < 8; ++r) {
    int m = mt*16 + r + 8*half, n = nt*16 + idx;
    float v = acc[r] + bias[n];
    if (RELU) v = fmaxf(v, 0.f);
    C[(size_t)m*Ndim + n] = v;
  }
}

// ---------------- fused: Wk generation + bkmo einsum + head MLP ----------------
// grid = (B/16, K), block = 512 (16 waves). Wave w owns batch b0+w in step 2.
__global__ void __launch_bounds__(512, 1)
fused_kernel(const float* __restrict__ item,
             const int*   __restrict__ mask,
             const float* __restrict__ sm,
             const float* __restrict__ Hf,     // (B,128) from ws
             const float* __restrict__ fcpW,   // (128, K*PPK)
             const float* __restrict__ fcpb,
             const float* __restrict__ w1, const float* __restrict__ b1,
             const float* __restrict__ w2, const float* __restrict__ b2,
             float* __restrict__ sk) {
  __shared__ unsigned short Hb_lds[16][128];      // bf16 bits, [b][c]
  __shared__ float          bk_lds[16][Om];
  __shared__ unsigned short wk_lds[16][Om][DC];   // bf16 bits, [b][o][d_local]
  __shared__ float          avg_lds[16][Om];

  const int tid = threadIdx.x, wave = tid >> 5, lane = tid & 31;
  const int half = lane >> 4, idx = lane & 15;
  const int b0 = blockIdx.x * 16, k = blockIdx.y;
  const size_t kbase = (size_t)k * PPK;

  // stage H tile (f32 -> bf16 bits)
  for (int i = tid; i < 16*128; i += 512) {
    int b = i >> 7, c = i & 127;
    Hb_lds[b][c] = bf16b(Hf[(size_t)(b0 + b)*128 + c]);
  }
  // per-batch generated bias bk[b][o] = H[b]@Fbk + fcp_b slice (f32)
  for (int i = tid; i < 16*Om; i += 512) {
    int b = i >> 6, o = i & 63;
    float s = fcpb[kbase + (size_t)Dm*Om + o];
    const float* hrow = Hf + (size_t)(b0 + b)*128;
    for (int c = 0; c < 128; ++c)
      s += hrow[c] * fcpW[(size_t)c*NROW + kbase + (size_t)Dm*Om + o];
    bk_lds[b][o] = s;
  }
  __syncthreads();

  // A fragments for step 1 (Hb 16x128, chunk-invariant): M=b, K=c.
  v16bf afr[4];
#pragma unroll
  for (int kk = 0; kk < 4; ++kk) {
    v8u p;
#pragma unroll
    for (int j = 0; j < 8; ++j) {
      int e = 2*j;
      int c = kk*32 + e + 8*half + ((e >= 8) ? 8 : 0);  // even
      p[j] = *(const unsigned*)&Hb_lds[idx][c];
    }
    afr[kk] = __builtin_bit_cast(v16bf, p);
  }

  v8f acc2[4];
#pragma unroll
  for (int ot = 0; ot < 4; ++ot)
#pragma unroll
    for (int r = 0; r < 8; ++r) acc2[ot][r] = 0.f;

#pragma unroll 1
  for (int ch = 0; ch < Dm/DC; ++ch) {
    const int d0 = ch * DC;
    // --- step 1: Wk_chunk[16b][DC d][64 o] = Hb(16x128) @ Fk_chunk ---
    // wave w covers d_local = w. kk outer so the 8 strided row addresses are
    // computed once and reused across the 4 o-subtiles (column offset folds
    // into the instruction offset).
    const float* colp = fcpW + kbase + (size_t)(d0 + wave)*Om + idx;
    v8f acc[4];
#pragma unroll
    for (int nt = 0; nt < 4; ++nt)
#pragma unroll
      for (int r = 0; r < 8; ++r) acc[nt][r] = 0.f;
#pragma unroll
    for (int kk = 0; kk < 4; ++kk) {
#pragma unroll
      for (int nt = 0; nt < 4; ++nt) {
        int o0 = nt * 16;
        v8u p;
#pragma unroll
        for (int j = 0; j < 8; ++j) {
          int c = kk*32 + 2*j + 8*half + ((j >= 4) ? 8 : 0);
          p[j] = pk_bf16(colp[(size_t)c*NROW + o0], colp[(size_t)(c + 1)*NROW + o0]);
        }
        acc[nt] = __builtin_amdgcn_wmma_f32_16x16x32_bf16(false, afr[kk],
                                                          false, __builtin_bit_cast(v16bf, p),
                                                          (short)0, acc[nt], false, false);
      }
    }
#pragma unroll
    for (int nt = 0; nt < 4; ++nt)
#pragma unroll
      for (int r = 0; r < 8; ++r)
        wk_lds[r + 8*half][nt*16 + idx][wave] = bf16b(acc[nt][r]);
    __syncthreads();

    // --- step 2: ep_pre[b=wave] += e(16m x 16d, zero-padded to K=32) @ Wk(16d x 64o)
    v8u pa;
    {
      const float* ip = item + ((size_t)(b0 + wave)*Mm + idx)*Dm;
      const float* sp = sm + (size_t)k*Dm;
#pragma unroll
      for (int j = 0; j < 8; ++j) {
        int e = 2*j;
        int dk = e + 8*half + ((e >= 8) ? 8 : 0);
        if (dk < DC) {
          int d = d0 + dk;
          float2 iv = ld2(&ip[d]);
          float2 sv = ld2(&sp[d]);
          pa[j] = pk_bf16(iv.x*sv.x, iv.y*sv.y);
        } else pa[j] = 0u;
      }
    }
    v16bf aef = __builtin_bit_cast(v16bf, pa);
#pragma unroll
    for (int ot = 0; ot < 4; ++ot) {
      v8u p;
#pragma unroll
      for (int j = 0; j < 8; ++j) {
        int dk = 2*j + 8*half + ((j >= 4) ? 8 : 0);
        p[j] = (dk < DC) ? *(const unsigned*)&wk_lds[wave][ot*16 + idx][dk] : 0u;
      }
      acc2[ot] = __builtin_amdgcn_wmma_f32_16x16x32_bf16(false, aef,
                                                         false, __builtin_bit_cast(v16bf, p),
                                                         (short)0, acc2[ot], false, false);
    }
    __syncthreads();
  }

  // --- finalize: bias + relu + masked mean over m ---
  const int b = b0 + wave;
  float cnt = 0.f;
#pragma unroll
  for (int m = 0; m < Mm; ++m) cnt += (float)mask[b*Mm + m];
  cnt = fmaxf(cnt, 1e-6f);
#pragma unroll
  for (int ot = 0; ot < 4; ++ot) {
    int o = ot*16 + idx;
    float s = 0.f;
#pragma unroll
    for (int r = 0; r < 8; ++r) {
      int m = r + 8*half;
      float v = fmaxf(acc2[ot][r] + bk_lds[wave][o], 0.f);
      s += (float)mask[b*Mm + m] * v;
    }
    s += __shfl_xor(s, 16, 32);
    avg_lds[wave][o] = s / cnt;
  }
  __syncthreads();

  // --- tiny per-k MLP head: t = relu(avg@W1+b1); sk = t@W2 + b2 ---
  float sp2 = 0.f;
#pragma unroll
  for (int j = 0; j < 4; ++j) {
    int h = lane + 32*j;
    float t = b1[(size_t)k*Hh + h];
#pragma unroll
    for (int o = 0; o < Om; ++o)
      t += avg_lds[wave][o] * w1[((size_t)k*Om + o)*Hh + h];
    t = fmaxf(t, 0.f);
    sp2 += t * w2[(size_t)k*Hh + h];
  }
#pragma unroll
  for (int off = 16; off >= 1; off >>= 1) sp2 += __shfl_xor(sp2, off, 32);
  if (lane == 0) sk[(size_t)k*Bm + b] = sp2 + b2[k];
}

// ---------------- mean over K + sigmoid ----------------
__global__ void finalize_kernel(const float* __restrict__ sk, float* __restrict__ out) {
  int b = blockIdx.x * blockDim.x + threadIdx.x;
  if (b < Bm) {
    float s = 0.f;
#pragma unroll
    for (int k = 0; k < Kk; ++k) s += sk[(size_t)k*Bm + b];
    s *= (1.f / (float)Kk);
    out[b] = 1.f / (1.f + expf(-s));
  }
}

extern "C" void kernel_launch(void* const* d_in, const int* in_sizes, int n_in,
                              void* d_out, int out_size, void* d_ws, size_t ws_size,
                              hipStream_t stream) {
  const float* item  = (const float*)d_in[0];
  const int*   mask  = (const int*)  d_in[1];
  const float* sm    = (const float*)d_in[2];
  const float* ceW1  = (const float*)d_in[3];
  const float* ceb1  = (const float*)d_in[4];
  const float* ceW2  = (const float*)d_in[5];
  const float* ceb2  = (const float*)d_in[6];
  const float* ceW3  = (const float*)d_in[7];
  const float* ceb3  = (const float*)d_in[8];
  const float* fcpW  = (const float*)d_in[9];
  const float* fcpb  = (const float*)d_in[10];
  const float* w1    = (const float*)d_in[11];
  const float* b1    = (const float*)d_in[12];
  const float* w2    = (const float*)d_in[13];
  const float* b2    = (const float*)d_in[14];
  float* out = (float*)d_out;

  float* ws  = (float*)d_ws;
  float* ctx = ws;                 // 512*512
  float* h1  = ctx + 262144;       // 512*512
  float* h2  = h1  + 262144;       // 512*256
  float* Hf  = h2  + 131072;       // 512*128
  float* sk  = Hf  + 65536;        // 8*512

  ctx_kernel<<<Bm, 256, 0, stream>>>(item, mask, ctx);
  gemm_wmma<1><<<128, 256, 0, stream>>>(ctx, ceW1, ceb1, h1, 512, 512, 512);
  gemm_wmma<1><<< 64, 256, 0, stream>>>(h1,  ceW2, ceb2, h2, 512, 256, 512);
  gemm_wmma<0><<< 32, 256, 0, stream>>>(h2,  ceW3, ceb3, Hf, 512, 128, 256);
  fused_kernel<<<dim3(Bm/16, Kk), 512, 0, stream>>>(item, mask, sm, Hf, fcpW, fcpb,
                                                    w1, b1, w2, b2, sk);
  finalize_kernel<<<(Bm + 255)/256, 256, 0, stream>>>(sk, out);
}
